// MultiHeadedAttention_47210280518045
// MI455X (gfx1250) — matrix operand order repaired and checked by reading, not compiled, not run on hardware
//
#include <hip/hip_runtime.h>
#include <cstdint>
#include <cstddef>

// ---------------- types ----------------
typedef _Float16 h4  __attribute__((ext_vector_type(4)));
typedef _Float16 h8  __attribute__((ext_vector_type(8)));
typedef _Float16 v16h __attribute__((ext_vector_type(16)));
typedef float    v8f  __attribute__((ext_vector_type(8)));
typedef float    f32x4 __attribute__((ext_vector_type(4)));
typedef int      v4i  __attribute__((ext_vector_type(4)));

#define HH     16
#define DK     64
#define DMODEL 1024
#define BB     4
#define SS     2048

// Async global->LDS copy path (CDNA5 GLOBAL_LOAD_ASYNC_TO_LDS_B128 + ASYNCcnt),
// guarded so host pass / older toolchains fall back to load+ds_store staging.
#if defined(__gfx1250__) && __has_builtin(__builtin_amdgcn_global_load_async_to_lds_b128) && __has_builtin(__builtin_amdgcn_s_wait_asynccnt)
#define USE_ASYNC_LDS 1
#else
#define USE_ASYNC_LDS 0
#endif

static __device__ inline v16h pack16(h8 lo, h8 hi) {
  v16h r;
#pragma unroll
  for (int i = 0; i < 8; ++i) { r[i] = lo[i]; r[8 + i] = hi[i]; }
  return r;
}

static __device__ inline void copy16_g2l(const _Float16* g, _Float16* l) {
#if USE_ASYNC_LDS
  // builtin expects (global int4*, shared int4*, imm offset, imm cpol)
  __builtin_amdgcn_global_load_async_to_lds_b128(
      (__attribute__((address_space(1))) v4i*)g,
      (__attribute__((address_space(3))) v4i*)l, 0, 0);
#else
  *(h8*)l = *(const h8*)g;
#endif
}

static __device__ inline void wait_async_all() {
#if USE_ASYNC_LDS
  __builtin_amdgcn_s_wait_asynccnt(0);
#endif
}

// =====================================================================
// Tiled WMMA GEMM:  C(f32) = A(f32, MxK row-major) * B(f32, KxN row-major) + bias
// f32 -> f16 conversion while staging into LDS (async copy not usable: convert).
// mode 0: outF[row*N+col] f32
// mode 1: outH head-split [b][h][s][d] f16   (Q, K)
// mode 2: outH head-split [b][h][d][s] f16   (V, d-major)
// Block: 128 threads (4 waves), tile 128x128, BK=32.
// Wave (wm,wn in 0..1) owns a 64x64 sub-tile -> 4 A-frags + 4 B-frags, 16 WMMAs.
// =====================================================================
__global__ __launch_bounds__(128) void gemm_f16_wmma(
    const float* __restrict__ A, const float* __restrict__ Bw,
    const float* __restrict__ bias,
    float* __restrict__ outF, _Float16* __restrict__ outH,
    int M, int N, int K, int mode)
{
  __shared__ _Float16 Alds[128][40];   // [m][k], 80B rows (16B aligned)
  __shared__ _Float16 Blds[128][40];   // [n][k] (transposed while loading)

  const int t    = threadIdx.x;
  const int lane = t & 31;
  const int wave = t >> 5;
  const int wm   = wave >> 1;          // 0..1
  const int wn   = wave & 1;           // 0..1
  const int lr   = lane & 15;
  const int hs   = lane >> 4;          // lane half select
  const int blockRow = blockIdx.y * 128;
  const int blockCol = blockIdx.x * 128;

  v8f acc[4][4];
#pragma unroll
  for (int mi = 0; mi < 4; ++mi)
#pragma unroll
    for (int ni = 0; ni < 4; ++ni)
#pragma unroll
      for (int e = 0; e < 8; ++e) acc[mi][ni][e] = 0.0f;

  for (int k0 = 0; k0 < K; k0 += 32) {
    // ---- stage A tile (128x32) as f16 ----
    {
      const int rowb = t >> 3;            // 0..15
      const int cg   = (t & 7) * 4;       // k group 0,4,...,28
#pragma unroll
      for (int p = 0; p < 8; ++p) {
        const int row = p * 16 + rowb;
        f32x4 v = *(const f32x4*)(A + (size_t)(blockRow + row) * K + k0 + cg);
        h4 hv;
#pragma unroll
        for (int i = 0; i < 4; ++i) hv[i] = (_Float16)v[i];
        *(h4*)(&Alds[row][cg]) = hv;      // ds_store_b64
      }
    }
    // ---- stage B tile (32x128) transposed as f16 -> Blds[n][k] ----
    {
      const int kk = t >> 5;              // 0..3
      const int ng = lane * 4;            // 0..124
#pragma unroll
      for (int p = 0; p < 8; ++p) {
        const int k = p * 4 + kk;
        f32x4 v = *(const f32x4*)(Bw + (size_t)(k0 + k) * N + blockCol + ng);
#pragma unroll
        for (int i = 0; i < 4; ++i) Blds[ng + i][k] = (_Float16)v[i];
      }
    }
    __syncthreads();

    // ---- A fragments (16x32): lanes 0-15 row M=lr K=0..15;
    //      lanes 16-31 K=16..31 (matches B fragments) ----
    v16h afrag[4];
#pragma unroll
    for (int mi = 0; mi < 4; ++mi) {
      const int row = wm * 64 + mi * 16 + lr;
      const int o0 = hs * 16;
      const int o1 = hs * 16 + 8;
      h8 lo = *(const h8*)(&Alds[row][o0]);   // ds_load_b128
      h8 hi = *(const h8*)(&Alds[row][o1]);
      afrag[mi] = pack16(lo, hi);
    }
    // ---- B fragments (32x16): lanes 0-15 N=lr K=0..15; lanes 16-31 K=16..31 ----
    v16h bfrag[4];
#pragma unroll
    for (int ni = 0; ni < 4; ++ni) {
      const int col = wn * 64 + ni * 16 + lr;
      const int o   = hs * 16;
      h8 lo = *(const h8*)(&Blds[col][o]);
      h8 hi = *(const h8*)(&Blds[col][o + 8]);
      bfrag[ni] = pack16(lo, hi);
    }
#pragma unroll
    for (int mi = 0; mi < 4; ++mi)
#pragma unroll
      for (int ni = 0; ni < 4; ++ni)
        acc[mi][ni] = __builtin_amdgcn_wmma_f32_16x16x32_f16(
            false, afrag[mi], false, bfrag[ni], (short)0, acc[mi][ni], false, false);
    __syncthreads();
  }

  // ---- epilogue: C layout = VGPR vi -> M=vi / vi+8 by lane half, N=lane&15 ----
#pragma unroll
  for (int mi = 0; mi < 4; ++mi) {
#pragma unroll
    for (int ni = 0; ni < 4; ++ni) {
      const int col = blockCol + wn * 64 + ni * 16 + lr;
      const float bv = bias[col];
#pragma unroll
      for (int vi = 0; vi < 8; ++vi) {
        const int row = blockRow + wm * 64 + mi * 16 + vi + 8 * hs;
        const float r = acc[mi][ni][vi] + bv;
        if (mode == 0) {
          outF[(size_t)row * N + col] = r;
        } else {
          const int b = row >> 11;       // row / S
          const int s = row & (SS - 1);
          const int h = col >> 6;        // col / DK
          const int d = col & (DK - 1);
          if (mode == 1)
            outH[(((size_t)b * HH + h) * SS + s) * DK + d] = (_Float16)r;
          else
            outH[(((size_t)b * HH + h) * DK + d) * SS + s] = (_Float16)r;
        }
      }
    }
  }
}

// =====================================================================
// Flash attention:  out[b,s,h*64+d] = softmax(Q K^T / 8, mask) V
// Qh,Kh: f16 [B][H][S][64];  Vt: f16 [B][H][64][S];  mask: bool [B][S][S]
// 128 threads = 4 waves sharing one (b,h); each wave owns 16 query rows.
// KV tiles of 64 keys staged once per block into double-buffered LDS via
// async global->LDS copies (shared by all 4 waves: 4x fewer global requests,
// copies overlap compute, ASYNCcnt + barrier for synchronization).
// Per tile per wave: 8 score WMMAs + 8 PV WMMAs.
// =====================================================================
__global__ __launch_bounds__(128) void attn_flash_wmma(
    const _Float16* __restrict__ Qh, const _Float16* __restrict__ Kh,
    const _Float16* __restrict__ Vt, const unsigned char* __restrict__ mask,
    float* __restrict__ out)
{
  __shared__ _Float16 Klds[2][64 * DK];    // [key][d], 128B rows
  __shared__ _Float16 Vlds[2][DK * 64];    // [d][key], 128B rows
  __shared__ _Float16 Plds[4][16][72];     // per-wave 16x64 P tile, 144B rows

  const int t    = threadIdx.x;
  const int lane = t & 31;
  const int wave = t >> 5;
  const int lr   = lane & 15;
  const int hs   = lane >> 4;

  const int b  = blockIdx.z;
  const int h  = blockIdx.y;
  const int q0 = blockIdx.x * 64 + wave * 16;

  const _Float16* Qbase = Qh + (((size_t)b * HH + h) * SS) * DK;
  const _Float16* Kbase = Kh + (((size_t)b * HH + h) * SS) * DK;
  const _Float16* Vbase = Vt + (((size_t)b * HH + h) * DK) * SS;
  const unsigned char* Mbase = mask + ((size_t)b * SS) * SS;

  // ---- Q A-fragments (per wave, loaded once): chunk c covers d = 32c..+31 ----
  v16h qf[2];
  {
    const _Float16* qp = Qbase + (size_t)(q0 + lr) * DK;
#pragma unroll
    for (int c = 0; c < 2; ++c) {
      const int o0 = c * 32 + hs * 16;
      const int o1 = c * 32 + hs * 16 + 8;
      h8 lo = *(const h8*)(qp + o0);   // global_load_b128
      h8 hi = *(const h8*)(qp + o1);
      qf[c] = pack16(lo, hi);
    }
  }

  // cooperative KV tile staging (128 threads, 8 x 16B chunks each per tile)
  auto stage_tile = [&](int kv0, int buf) {
    const _Float16* kg = Kbase + (size_t)kv0 * DK;   // 8KB contiguous
#pragma unroll
    for (int p = 0; p < 4; ++p) {
      const int c = t + p * 128;                     // 16B chunk id
      copy16_g2l(kg + c * 8, &Klds[buf][c * 8]);
    }
#pragma unroll
    for (int p = 0; p < 4; ++p) {
      const int c  = t + p * 128;
      const int d  = c >> 3;
      const int of = (c & 7) * 8;
      copy16_g2l(Vbase + (size_t)d * SS + kv0 + of, &Vlds[buf][d * 64 + of]);
    }
  };

  float mrow[8], lrow[8];
  v8f oacc[4];
#pragma unroll
  for (int vi = 0; vi < 8; ++vi) { mrow[vi] = -1e30f; lrow[vi] = 0.0f; }
#pragma unroll
  for (int j = 0; j < 4; ++j)
#pragma unroll
    for (int e = 0; e < 8; ++e) oacc[j][e] = 0.0f;

  stage_tile(0, 0);
  wait_async_all();
  __syncthreads();

  for (int kv0 = 0; kv0 < SS; kv0 += 64) {
    const int buf = (kv0 >> 6) & 1;
    // prefetch next tile into the other buffer (overlaps with compute below)
    if (kv0 + 64 < SS) stage_tile(kv0 + 64, buf ^ 1);

    // ---- K^T B-fragments from LDS: half selects keys kv0+16*half..+15 ----
    v16h kb[4][2];
#pragma unroll
    for (int half = 0; half < 4; ++half) {
      const _Float16* kp = &Klds[buf][(half * 16 + lr) * DK];
#pragma unroll
      for (int c = 0; c < 2; ++c) {
        const int o = c * 32 + hs * 16;
        h8 lo = *(const h8*)(kp + o);          // ds_load_b128
        h8 hi = *(const h8*)(kp + o + 8);
        kb[half][c] = pack16(lo, hi);
      }
    }
    // ---- scores: 16q x 64k via 8 WMMAs ----
    v8f s[4];
#pragma unroll
    for (int half = 0; half < 4; ++half) {
#pragma unroll
      for (int e = 0; e < 8; ++e) s[half][e] = 0.0f;
      s[half] = __builtin_amdgcn_wmma_f32_16x16x32_f16(
          false, qf[0], false, kb[half][0], (short)0, s[half], false, false);
      s[half] = __builtin_amdgcn_wmma_f32_16x16x32_f16(
          false, qf[1], false, kb[half][1], (short)0, s[half], false, false);
    }

    // ---- online softmax per row (row vi lives across 16 lanes of one VGPR) ----
#pragma unroll
    for (int vi = 0; vi < 8; ++vi) {
      const int qr = q0 + vi + 8 * hs;
      const unsigned char* mp = Mbase + (size_t)qr * SS + kv0 + lr;
      float a[4];
#pragma unroll
      for (int half = 0; half < 4; ++half) {
        a[half] = s[half][vi] * 0.125f;        // 1/sqrt(64)
        if (!mp[half * 16]) a[half] = -1e9f;
      }
      float tmax = fmaxf(fmaxf(a[0], a[1]), fmaxf(a[2], a[3]));
      tmax = fmaxf(tmax, __shfl_xor(tmax, 1));
      tmax = fmaxf(tmax, __shfl_xor(tmax, 2));
      tmax = fmaxf(tmax, __shfl_xor(tmax, 4));
      tmax = fmaxf(tmax, __shfl_xor(tmax, 8));
      const float mnew  = fmaxf(mrow[vi], tmax);
      const float alpha = __expf(mrow[vi] - mnew);
      float p[4], ps = 0.0f;
#pragma unroll
      for (int half = 0; half < 4; ++half) { p[half] = __expf(a[half] - mnew); ps += p[half]; }
      ps += __shfl_xor(ps, 1);
      ps += __shfl_xor(ps, 2);
      ps += __shfl_xor(ps, 4);
      ps += __shfl_xor(ps, 8);
      lrow[vi] = lrow[vi] * alpha + ps;
      mrow[vi] = mnew;
#pragma unroll
      for (int j = 0; j < 4; ++j) oacc[j][vi] *= alpha;
      // C-layout -> LDS (row-major 16x64 P tile) for A-layout reload
      const int prow = vi + 8 * hs;
#pragma unroll
      for (int half = 0; half < 4; ++half)
        Plds[wave][prow][lr + half * 16] = (_Float16)p[half];
    }

    // ---- P as A-fragments (intra-wave LDS RAW; DS ops in-order per wave) ----
    v16h pf[2];
#pragma unroll
    for (int pc = 0; pc < 2; ++pc) {
      const int o0 = pc * 32 + hs * 16;
      const int o1 = pc * 32 + hs * 16 + 8;
      h8 lo = *(const h8*)(&Plds[wave][lr][o0]);
      h8 hi = *(const h8*)(&Plds[wave][lr][o1]);
      pf[pc] = pack16(lo, hi);
    }

    // ---- V B-fragments from LDS (d-major: contiguous in key dim) ----
#pragma unroll
    for (int pc = 0; pc < 2; ++pc) {
#pragma unroll
      for (int j = 0; j < 4; ++j) {
        const int d = j * 16 + lr;
        const _Float16* vp = &Vlds[buf][d * 64 + pc * 32 + hs * 16];
        h8 lo = *(const h8*)(vp);
        h8 hi = *(const h8*)(vp + 8);
        v16h vb = pack16(lo, hi);
        oacc[j] = __builtin_amdgcn_wmma_f32_16x16x32_f16(
            false, pf[pc], false, vb, (short)0, oacc[j], false, false);
      }
    }

    // next tile's copies must be complete (and this tile's reads done in all
    // waves) before buffers flip / get overwritten
    wait_async_all();
    __syncthreads();
  }

  // ---- epilogue: normalize by row sum, write f32 [B,S,H*64] ----
#pragma unroll
  for (int j = 0; j < 4; ++j) {
    const int d = j * 16 + lr;
#pragma unroll
    for (int vi = 0; vi < 8; ++vi) {
      const int qr = q0 + vi + 8 * hs;
      out[((size_t)b * SS + qr) * DMODEL + h * DK + d] = oacc[j][vi] / lrow[vi];
    }
  }
}

// =====================================================================
// Host launcher
// =====================================================================
extern "C" void kernel_launch(void* const* d_in, const int* in_sizes, int n_in,
                              void* d_out, int out_size, void* d_ws, size_t ws_size,
                              hipStream_t stream) {
  const float* query = (const float*)d_in[0];
  const float* key_  = (const float*)d_in[1];
  const float* value = (const float*)d_in[2];
  const unsigned char* mask = (const unsigned char*)d_in[3];
  const float* Wq = (const float*)d_in[4];
  const float* bq = (const float*)d_in[5];
  const float* Wk = (const float*)d_in[6];
  const float* bk = (const float*)d_in[7];
  const float* Wv = (const float*)d_in[8];
  const float* bv = (const float*)d_in[9];
  const float* Wo = (const float*)d_in[10];
  const float* bo = (const float*)d_in[11];
  float* out = (float*)d_out;

  // workspace: Qh(16MB) Kh(16MB) Vt(16MB) f16  +  attn(32MB) f32  = 80 MB
  char* ws = (char*)d_ws;
  _Float16* Qh  = (_Float16*)(ws);
  _Float16* Kh  = (_Float16*)(ws + ((size_t)16 << 20));
  _Float16* Vt  = (_Float16*)(ws + ((size_t)32 << 20));
  float*    att = (float*)   (ws + ((size_t)48 << 20));

  const dim3 gblk(DMODEL / 128, (BB * SS) / 128);  // (8, 64)
  const dim3 gthr(128);

  // Q/K/V projections (write head-split f16; V transposed to d-major)
  gemm_f16_wmma<<<gblk, gthr, 0, stream>>>(query, Wq, bq, nullptr, Qh, BB * SS, DMODEL, DMODEL, 1);
  gemm_f16_wmma<<<gblk, gthr, 0, stream>>>(key_,  Wk, bk, nullptr, Kh, BB * SS, DMODEL, DMODEL, 1);
  gemm_f16_wmma<<<gblk, gthr, 0, stream>>>(value, Wv, bv, nullptr, Vt, BB * SS, DMODEL, DMODEL, 2);

  // flash attention
  const dim3 ablk(SS / 64, HH, BB);                // (32, 16, 4)
  attn_flash_wmma<<<ablk, dim3(128), 0, stream>>>(Qh, Kh, Vt, mask, att);

  // output projection -> f32 d_out
  gemm_f16_wmma<<<gblk, gthr, 0, stream>>>(att, Wo, bo, out, nullptr, BB * SS, DMODEL, DMODEL, 0);
}
